// Geo2DEncoder_6390911336728
// MI455X (gfx1250) — matrix operand order repaired and checked by reading, not compiled, not run on hardware
//
#include <hip/hip_runtime.h>
#include <hip/hip_bf16.h>

typedef __attribute__((ext_vector_type(16))) _Float16 v16h;
typedef __attribute__((ext_vector_type(8)))  float    v8f;

#define FMAXV 3.402823466e38f

__device__ __forceinline__ float gelu_erf(float x) {
    return 0.5f * x * (1.0f + erff(x * 0.7071067811865476f));
}

// ---------------------------------------------------------------------------
// Weight prep: W [Kin,N] f32 row-major  ->  Wt [N,Kpad] f16 (transposed, K-padded)
// ---------------------------------------------------------------------------
__global__ void prep_wt_kernel(const float* __restrict__ W, _Float16* __restrict__ Wt,
                               int Kin, int N, int Kpad) {
    int idx = blockIdx.x * blockDim.x + threadIdx.x;
    int total = N * Kpad;
    if (idx >= total) return;
    int n = idx / Kpad;
    int k = idx % Kpad;
    float v = (k < Kin) ? W[k * N + n] : 0.0f;
    Wt[idx] = (_Float16)v;
}

// ---------------------------------------------------------------------------
// FPS
// ---------------------------------------------------------------------------
__global__ void fps_init_kernel(float* __restrict__ min_d, int N,
                                const float* __restrict__ P,
                                int* __restrict__ cidx, float* __restrict__ cxy) {
    int stride = gridDim.x * blockDim.x;
    for (int i = blockIdx.x * blockDim.x + threadIdx.x; i < N; i += stride)
        min_d[i] = FMAXV;
    if (blockIdx.x == 0 && threadIdx.x == 0) {
        cidx[0] = 0;
        cxy[0] = P[0];
        cxy[1] = P[1];
    }
}

__global__ void fps_update_kernel(const float* __restrict__ P, int N,
                                  const float* __restrict__ cxy, int s,
                                  float* __restrict__ min_d,
                                  float* __restrict__ pval, int* __restrict__ pidx) {
    __shared__ float sv[256];
    __shared__ int   si[256];
    const float2* __restrict__ P2 = (const float2*)P;
    int t = threadIdx.x;
    float cx = cxy[2 * (s - 1)];
    float cy = cxy[2 * (s - 1) + 1];
    float bv = -1.0f;
    int   bi = 0x7fffffff;
    int stride = gridDim.x * blockDim.x;
    for (int i = blockIdx.x * blockDim.x + t; i < N; i += stride) {
        __builtin_prefetch(&P2[i + stride], 0, 1);
        float2 p = P2[i];
        float dx = p.x - cx, dy = p.y - cy;
        float d = dx * dx + dy * dy;
        float md = min_d[i];
        if (d < md) { md = d; min_d[i] = md; }
        if (md > bv) { bv = md; bi = i; }
    }
    sv[t] = bv; si[t] = bi;
    __syncthreads();
    for (int st = 128; st > 0; st >>= 1) {
        if (t < st) {
            float v2 = sv[t + st]; int i2 = si[t + st];
            if (v2 > sv[t] || (v2 == sv[t] && i2 < si[t])) { sv[t] = v2; si[t] = i2; }
        }
        __syncthreads();
    }
    if (t == 0) { pval[blockIdx.x] = sv[0]; pidx[blockIdx.x] = si[0]; }
}

__global__ void fps_pick_kernel(const float* __restrict__ pval, const int* __restrict__ pidx,
                                const float* __restrict__ P,
                                int* __restrict__ cidx, float* __restrict__ cxy, int s) {
    __shared__ float sv[1024];
    __shared__ int   si[1024];
    int t = threadIdx.x;
    float v1 = pval[t];          int i1 = pidx[t];
    float v2 = pval[t + 1024];   int i2 = pidx[t + 1024];
    if (v2 > v1 || (v2 == v1 && i2 < i1)) { v1 = v2; i1 = i2; }
    sv[t] = v1; si[t] = i1;
    __syncthreads();
    for (int st = 512; st > 0; st >>= 1) {
        if (t < st) {
            float w = sv[t + st]; int j = si[t + st];
            if (w > sv[t] || (w == sv[t] && j < si[t])) { sv[t] = w; si[t] = j; }
        }
        __syncthreads();
    }
    if (t == 0) {
        int best = si[0];
        cidx[s] = best;
        cxy[2 * s]     = P[2 * best];
        cxy[2 * s + 1] = P[2 * best + 1];
    }
}

// ---------------------------------------------------------------------------
// kNN: one block per center; threshold + LDS candidate buffer + bitonic select
// ---------------------------------------------------------------------------
__device__ void knn_compact(float* cd, int* ci, int* cnt, float* thr, int t) {
    __syncthreads();
    int c = *cnt;
    for (int i = t; i < 2048; i += 1024) {
        if (i >= c) { cd[i] = FMAXV; ci[i] = 0x7fffffff; }
    }
    for (unsigned k = 2; k <= 2048u; k <<= 1) {
        for (unsigned j = k >> 1; j > 0; j >>= 1) {
            __syncthreads();
            for (unsigned idx = t; idx < 2048u; idx += 1024u) {
                unsigned l = idx ^ j;
                if (l > idx) {
                    float d1 = cd[idx], d2 = cd[l];
                    int   i1 = ci[idx], i2 = ci[l];
                    bool less = (d1 < d2) || (d1 == d2 && i1 < i2);
                    bool up = ((idx & k) == 0);
                    if (less != up) {
                        cd[idx] = d2; cd[l] = d1;
                        ci[idx] = i2; ci[l] = i1;
                    }
                }
            }
        }
    }
    __syncthreads();
    if (t == 0) {
        *cnt = (c < 64) ? c : 64;
        *thr = cd[63];
    }
    __syncthreads();
}

__global__ void knn_kernel(const float* __restrict__ P, int N,
                           const float* __restrict__ cxy, int* __restrict__ nidx) {
    __shared__ float cd[2048];
    __shared__ int   ci[2048];
    __shared__ int   cnt;
    __shared__ float thr;
    const float2* __restrict__ P2 = (const float2*)P;
    const int m = blockIdx.x;
    const int t = threadIdx.x;
    float cx = cxy[2 * m], cy = cxy[2 * m + 1];
    if (t == 0) { cnt = 0; thr = FMAXV; }
    __syncthreads();
    for (int base = 0; base < N; base += 1024) {
        int i = base + t;
        if (i < N) {
            __builtin_prefetch(&P2[i + 1024], 0, 1);
            float2 p = P2[i];
            float dx = p.x - cx, dy = p.y - cy;
            float d = dx * dx + dy * dy;
            if (d <= thr) {
                int pos = atomicAdd(&cnt, 1);
                cd[pos] = d;
                ci[pos] = i;
            }
        }
        __syncthreads();
        if (cnt > 1024) knn_compact(cd, ci, &cnt, &thr, t);
        __syncthreads();
    }
    knn_compact(cd, ci, &cnt, &thr, t);
    if (t < 64) nidx[m * 64 + t] = ci[t];
}

// ---------------------------------------------------------------------------
// Featurize: 29 geometric features (K-padded to 32), f16 for WMMA
// ---------------------------------------------------------------------------
__global__ void featurize_kernel(const float* __restrict__ P, const float* __restrict__ cxy,
                                 const int* __restrict__ nidx,
                                 _Float16* __restrict__ A1, float* __restrict__ rbuf) {
    int g = blockIdx.x * blockDim.x + threadIdx.x;
    if (g >= 1024) return;
    int m = g >> 6;
    int idx = nidx[g];
    float px = P[2 * idx], py = P[2 * idx + 1];
    float dx = px - cxy[2 * m], dy = py - cxy[2 * m + 1];
    float r = sqrtf(dx * dx + dy * dy);
    float th = atan2f(dy, dx);
    _Float16* row = A1 + (size_t)g * 32;
    row[0] = (_Float16)dx;
    row[1] = (_Float16)dy;
    row[2] = (_Float16)r;
    row[3] = (_Float16)sinf(th);
    row[4] = (_Float16)cosf(th);
    float f = 1.0f;
    #pragma unroll
    for (int l = 0; l < 6; ++l) {
        float ax = dx * f, ay = dy * f;
        row[5 + 4 * l + 0] = (_Float16)sinf(ax);
        row[5 + 4 * l + 1] = (_Float16)sinf(ay);
        row[5 + 4 * l + 2] = (_Float16)cosf(ax);
        row[5 + 4 * l + 3] = (_Float16)cosf(ay);
        f *= 2.0f;
    }
    row[29] = (_Float16)0.0f;
    row[30] = (_Float16)0.0f;
    row[31] = (_Float16)0.0f;
    rbuf[g] = r;
}

// ---------------------------------------------------------------------------
// WMMA GEMM, compile-time specialized: one wave32 per 16x16 output tile.
// A [ROWS,K] f16 row-major; Bt [N,K] f16 (transposed weights); f32 accumulate.
// out = act(A@B + bias) stored f16 (+ optional f32 copy for final output).
// K fully unrolled -> back-to-back v_wmma chains per tile.
// ---------------------------------------------------------------------------
template <int ROWS, int K, int N, int ACT>
__global__ void wmma_gemm_kernel(const _Float16* __restrict__ A,
                                 const _Float16* __restrict__ Bt,
                                 const float* __restrict__ bias,
                                 _Float16* __restrict__ outH,
                                 float* __restrict__ outF) {
    const int lane = threadIdx.x & 31;
    const int wave = blockIdx.x * (blockDim.x >> 5) + (threadIdx.x >> 5);
    constexpr int ntiles = N >> 4;
    constexpr int tiles = (ROWS >> 4) * ntiles;
    if (wave >= tiles) return;
    const int mtile = wave / ntiles;
    const int ntile = wave % ntiles;
    const int mn = lane & 15;        // row-in-tile for A; col-in-tile for B/C/D
    const int kh = lane >> 4;        // lane-half selects K sub-block

    const _Float16* Arow = A + (size_t)(mtile * 16 + mn) * K;
    const _Float16* Brow = Bt + (size_t)(ntile * 16 + mn) * K;

    union AV { v16h v; _Float16 h[16]; };
    v8f c = {0.f, 0.f, 0.f, 0.f, 0.f, 0.f, 0.f, 0.f};

    #pragma unroll
    for (int k0 = 0; k0 < K; k0 += 32) {
        AV a, b;
        #pragma unroll
        for (int e = 0; e < 16; ++e) {
            // A 16x32 f16 layout (ISA 7.12.2): lanes0-15 V0-3 -> K=0..7, V4-7 -> K=16..23;
            // lanes16-31 shifted by 8.
            int kA = (e < 8) ? (kh * 8 + e) : (16 + kh * 8 + (e - 8));
            a.h[e] = Arow[k0 + kA];
            // B 32x16 f16: same K order as A (lanes0-15 K=0..7,16..23; lanes16-31 shifted by 8).
            b.h[e] = Brow[k0 + kA];
        }
        c = __builtin_amdgcn_wmma_f32_16x16x32_f16(false, a.v, false, b.v,
                                                   (short)0, c, false, false);
    }

    union CV { v8f v; float f[8]; } cc;
    cc.v = c;
    const int col = ntile * 16 + mn;
    const float bv = bias[col];
    #pragma unroll
    for (int i = 0; i < 8; ++i) {
        int row = mtile * 16 + kh * 8 + i;   // C/D: VGPR i -> M=i (lanes0-15) / M=i+8
        float v = cc.f[i] + bv;
        if (ACT) v = gelu_erf(v);
        outH[(size_t)row * N + col] = (_Float16)v;
        if (outF) outF[(size_t)row * N + col] = v;
    }
}

// ---------------------------------------------------------------------------
// Pooling + stats -> z [16, 288] (h_max | h_mean | stats | zero-pad)
// ---------------------------------------------------------------------------
__global__ void pool_kernel(const _Float16* __restrict__ H, _Float16* __restrict__ zh) {
    int g = blockIdx.x * blockDim.x + threadIdx.x;
    if (g >= 2048) return;
    int m = g >> 7, ch = g & 127;
    float mx = -FMAXV, sum = 0.0f;
    for (int k = 0; k < 64; ++k) {
        float v = (float)H[(size_t)(m * 64 + k) * 128 + ch];
        mx = fmaxf(mx, v);
        sum += v;
    }
    zh[m * 288 + ch]       = (_Float16)mx;
    zh[m * 288 + 128 + ch] = (_Float16)(sum * (1.0f / 64.0f));
}

__global__ void stats_kernel(const float* __restrict__ rbuf, _Float16* __restrict__ zh) {
    int t = threadIdx.x;
    if (t >= 512) return;
    int m = t >> 5, cc = t & 31;
    float out = 0.0f;
    if (cc < 4) {
        const float* r = rbuf + m * 64;
        float mean = 0.0f, mx = -FMAXV, mn = FMAXV;
        for (int k = 0; k < 64; ++k) {
            float v = r[k];
            mean += v;
            mx = fmaxf(mx, v);
            mn = fminf(mn, v);
        }
        mean *= (1.0f / 64.0f);
        float var = 0.0f;
        for (int k = 0; k < 64; ++k) {
            float d = r[k] - mean;
            var += d * d;
        }
        var *= (1.0f / 63.0f);   // ddof=1
        float vals[4] = {mean, mx, mn, sqrtf(var)};
        out = vals[cc];
    }
    zh[m * 288 + 256 + cc] = (_Float16)out;
}

__global__ void copy_centers_kernel(const float* __restrict__ cxy, float* __restrict__ out) {
    int t = threadIdx.x;
    if (t < 32) out[t] = cxy[t];
}

// ---------------------------------------------------------------------------
// Launch
// ---------------------------------------------------------------------------
extern "C" void kernel_launch(void* const* d_in, const int* in_sizes, int n_in,
                              void* d_out, int out_size, void* d_ws, size_t ws_size,
                              hipStream_t stream) {
    const float* P  = (const float*)d_in[0];
    const float* W1 = (const float*)d_in[1];
    const float* b1 = (const float*)d_in[2];
    const float* W2 = (const float*)d_in[3];
    const float* b2 = (const float*)d_in[4];
    const float* W3 = (const float*)d_in[5];
    const float* b3 = (const float*)d_in[6];
    const float* W4 = (const float*)d_in[7];
    const float* b4 = (const float*)d_in[8];
    const int N = in_sizes[0] / 2;

    char* ws = (char*)d_ws;
    size_t off = 0;
    auto carve = [&](size_t bytes) -> char* {
        char* p = ws + off;
        off = (off + bytes + 255) & ~(size_t)255;
        return p;
    };
    float*    min_d = (float*)carve((size_t)N * 4);
    float*    pval  = (float*)carve(2048 * 4);
    int*      pidx  = (int*)carve(2048 * 4);
    int*      cidx  = (int*)carve(16 * 4);
    float*    cxy   = (float*)carve(32 * 4);
    int*      nidx  = (int*)carve(1024 * 4);
    float*    rbuf  = (float*)carve(1024 * 4);
    _Float16* A1h   = (_Float16*)carve((size_t)1024 * 32 * 2);
    _Float16* H1h   = (_Float16*)carve((size_t)1024 * 64 * 2);
    _Float16* Hh    = (_Float16*)carve((size_t)1024 * 128 * 2);
    _Float16* zh    = (_Float16*)carve((size_t)16 * 288 * 2);
    _Float16* Z3h   = (_Float16*)carve((size_t)16 * 128 * 2);
    _Float16* Z4h   = (_Float16*)carve((size_t)16 * 128 * 2);
    _Float16* W1t   = (_Float16*)carve((size_t)64 * 32 * 2);
    _Float16* W2t   = (_Float16*)carve((size_t)128 * 64 * 2);
    _Float16* W3t   = (_Float16*)carve((size_t)128 * 288 * 2);
    _Float16* W4t   = (_Float16*)carve((size_t)128 * 128 * 2);

    // Weight prep (transpose + pad + f16 convert)
    prep_wt_kernel<<<(64 * 32 + 255) / 256, 256, 0, stream>>>(W1, W1t, 29, 64, 32);
    prep_wt_kernel<<<(128 * 64 + 255) / 256, 256, 0, stream>>>(W2, W2t, 64, 128, 64);
    prep_wt_kernel<<<(128 * 288 + 255) / 256, 256, 0, stream>>>(W3, W3t, 260, 128, 288);
    prep_wt_kernel<<<(128 * 128 + 255) / 256, 256, 0, stream>>>(W4, W4t, 128, 128, 128);

    // FPS
    fps_init_kernel<<<2048, 256, 0, stream>>>(min_d, N, P, cidx, cxy);
    for (int s = 1; s < 16; ++s) {
        fps_update_kernel<<<2048, 256, 0, stream>>>(P, N, cxy, s, min_d, pval, pidx);
        fps_pick_kernel<<<1, 1024, 0, stream>>>(pval, pidx, P, cidx, cxy, s);
    }

    // kNN (16 centers, top-64 each)
    knn_kernel<<<16, 1024, 0, stream>>>(P, N, cxy, nidx);

    // Features
    featurize_kernel<<<4, 256, 0, stream>>>(P, cxy, nidx, A1h, rbuf);

    // MLP via WMMA: [1024,32]@[32,64] -> gelu -> [1024,64]@[64,128] -> gelu
    wmma_gemm_kernel<1024, 32, 64, 1><<<32, 256, 0, stream>>>(A1h, W1t, b1, H1h, nullptr);
    wmma_gemm_kernel<1024, 64, 128, 1><<<64, 256, 0, stream>>>(H1h, W2t, b2, Hh, nullptr);

    // Pool + stats -> z [16,288]
    pool_kernel<<<8, 256, 0, stream>>>(Hh, zh);
    stats_kernel<<<1, 512, 0, stream>>>(rbuf, zh);

    // Head: [16,288]@[288,128] -> gelu -> [16,128]@[128,128] + b4 -> F (f32)
    wmma_gemm_kernel<16, 288, 128, 1><<<1, 256, 0, stream>>>(zh, W3t, b3, Z3h, nullptr);
    wmma_gemm_kernel<16, 128, 128, 0><<<1, 256, 0, stream>>>(Z3h, W4t, b4, Z4h, (float*)d_out);

    // Centers -> d_out tail
    copy_centers_kernel<<<1, 32, 0, stream>>>(cxy, (float*)d_out + 2048);
}
